// Attention_12987981103467
// MI455X (gfx1250) — matrix operand order repaired and checked by reading, not compiled, not run on hardware
//
#include <hip/hip_runtime.h>
#include <hip/hip_bf16.h>

typedef __attribute__((ext_vector_type(16))) __bf16 v16bf;
typedef __attribute__((ext_vector_type(8)))  float  v8f;

typedef union { unsigned int u[8]; v16bf v; } FragU;

#define EMB   2048
#define NQ    32
#define NKV   8
#define NREP  4
#define HEAD  64
#define BB    2
#define TT    2048
#define MTOT  (BB*TT)       // 4096 rows
#define KVDIM (NKV*HEAD)    // 512

// A-fragment (16x32 bf16) dword j -> K offset, contiguous order matching B:
// lanes 0..15: K=0..15, lanes 16..31: K=16..31
__device__ __forceinline__ int a_koff(int j, int hi) {
  return hi * 16 + 2 * j;
}

// ---------------------------------------------------------------- casts
__global__ void cast_f32_bf16(const float* __restrict__ in,
                              __bf16* __restrict__ out, int n) {
  int i = blockIdx.x * blockDim.x + threadIdx.x;
  if (i < n) out[i] = (__bf16)in[i];
}

// ---------------------------------------------------------------- GEMM
// C[M,N] (f32) = A[M,K](bf16) * B[K,N](bf16) + bias[N](f32)
// block tile 128x64, 8 waves of 32x32 (2x2 WMMA frags), K step 32.
#define GBM 128
#define GBN 64
#define LSTR 34   // padded LDS row stride (bf16 elems): 68B = 17 banks, conflict-free

__global__ __launch_bounds__(256) void gemm_bias_bf16(
    const __bf16* __restrict__ A, const __bf16* __restrict__ Bm,
    const float* __restrict__ bias, float* __restrict__ C,
    int M, int N, int K)
{
  __shared__ __bf16 As[GBM * LSTR];   // [128][34] row-major (M x K)
  __shared__ __bf16 Bs[GBN * LSTR];   // [64][34] transposed  (N x K)
  const int tid  = threadIdx.x;
  const int lane = tid & 31;
  const int wave = tid >> 5;
  const int n16  = lane & 15;
  const int hi   = lane >> 4;
  const int m0 = blockIdx.y * GBM;
  const int n0 = blockIdx.x * GBN;
  const int mw = (wave >> 1) * 32;
  const int nw = (wave & 1) * 32;

  v8f acc[2][2];
  for (int ii = 0; ii < 2; ++ii)
    for (int jj = 0; jj < 2; ++jj)
      for (int r = 0; r < 8; ++r) acc[ii][jj][r] = 0.0f;

  for (int kk = 0; kk < K; kk += 32) {
    // stage A: 128x32 bf16 = 2048 dwords, 8 per thread (coalesced)
    #pragma unroll
    for (int i = 0; i < 8; ++i) {
      int lin = tid + i * 256;
      int row = lin >> 4;            // 16 dwords per row
      int kc  = (lin & 15) << 1;
      unsigned int d = *(const unsigned int*)(A + (size_t)(m0 + row) * K + kk + kc);
      *(unsigned int*)&As[row * LSTR + kc] = d;
    }
    // stage B transposed: 32x64 bf16 = 1024 dwords, 4 per thread
    #pragma unroll
    for (int i = 0; i < 4; ++i) {
      int lin = tid + i * 256;
      int kr = lin >> 5;             // 32 dwords per K row
      int nc = (lin & 31) << 1;
      const __bf16* gp = Bm + (size_t)(kk + kr) * N + n0 + nc;
      Bs[nc * LSTR + kr]       = gp[0];
      Bs[(nc + 1) * LSTR + kr] = gp[1];
    }
    __syncthreads();

    FragU afr[2], bfr[2];
    #pragma unroll
    for (int mi = 0; mi < 2; ++mi)
      #pragma unroll
      for (int j = 0; j < 8; ++j)
        afr[mi].u[j] = *(const unsigned int*)&As[(mw + mi * 16 + n16) * LSTR + a_koff(j, hi)];
    #pragma unroll
    for (int ni = 0; ni < 2; ++ni)
      #pragma unroll
      for (int j = 0; j < 8; ++j)
        bfr[ni].u[j] = *(const unsigned int*)&Bs[(nw + ni * 16 + n16) * LSTR + hi * 16 + 2 * j];

    #pragma unroll
    for (int mi = 0; mi < 2; ++mi)
      #pragma unroll
      for (int ni = 0; ni < 2; ++ni)
        acc[mi][ni] = __builtin_amdgcn_wmma_f32_16x16x32_bf16(
            false, afr[mi].v, false, bfr[ni].v, (short)0, acc[mi][ni], false, false);
    __syncthreads();
  }

  #pragma unroll
  for (int mi = 0; mi < 2; ++mi)
    #pragma unroll
    for (int ni = 0; ni < 2; ++ni)
      #pragma unroll
      for (int r = 0; r < 8; ++r) {
        int row = m0 + mw + mi * 16 + r + 8 * hi;
        int col = n0 + nw + ni * 16 + n16;
        C[(size_t)row * N + col] = acc[mi][ni][r] + bias[col];
      }
}

// ------------------------------------------------------- RoPE + relayout
// in : f32 [B, T, H, 64]   out: bf16 [B, H, T, 64]
__global__ void rope_to_heads(const float* __restrict__ in,
                              __bf16* __restrict__ out, int H, int total)
{
  int i = blockIdx.x * blockDim.x + threadIdx.x;
  if (i >= total) return;
  int d = i & 63;
  int rest = i >> 6;
  int h = rest % H;
  int q = rest / H;
  int t = q % TT;
  int b = q / TT;
  int dr = d & 31;
  // inv_freq = 10000^(-dr/32) = exp2(-dr * log2(10000)/32)
  float freq = exp2f(-(float)dr * 0.41524101186092029f);
  float ang = (float)t * freq;
  float c = cosf(ang), s = sinf(ang);
  float v  = in[i];
  float v2 = (d < 32) ? in[i + 32] : in[i - 32];
  float vo = (d < 32) ? (v * c - v2 * s) : (v * c + v2 * s);
  out[(((size_t)b * H + h) * TT + t) * 64 + d] = (__bf16)vo;
}

// in : f32 [B, T, NKV, 64]  out: bf16 [B, NKV, 64, T]  (transposed for PV B-frags)
__global__ void v_cast_T(const float* __restrict__ in,
                         __bf16* __restrict__ out, int total)
{
  int i = blockIdx.x * blockDim.x + threadIdx.x;
  if (i >= total) return;
  int d  = i & 63;
  int kv = (i >> 6) & 7;
  int t  = (i >> 9) & (TT - 1);
  int b  = i >> 20;
  out[(((size_t)b * NKV + kv) * HEAD + d) * TT + t] = (__bf16)in[i];
}

// --------------------------------------------------------- flash attention
// One wave per 16-row Q tile; K streamed in chunks of 32 keys.
// Q: [B,NQ,T,64]  K: [B,NKV,T,64]  Vt: [B,NKV,64,T]  Y: [B,T,NQ*64] (all bf16)
#define PSTR 34

__global__ __launch_bounds__(256) void flash_attn(
    const __bf16* __restrict__ Qb, const __bf16* __restrict__ Kb,
    const __bf16* __restrict__ Vt, __bf16* __restrict__ Y)
{
  __shared__ __bf16 Pb[8 * 16 * PSTR];          // per-wave P staging
  const int lane = threadIdx.x & 31;
  const int wave = threadIdx.x >> 5;
  const int n16 = lane & 15, hi = lane >> 4;
  const int b = blockIdx.z, h = blockIdx.y;
  const int kvh = h >> 2;                       // NREP = 4
  const int q0 = (blockIdx.x * 8 + wave) * 16;
  const __bf16* Q  = Qb + ((size_t)(b * NQ  + h)   * TT) * HEAD;
  const __bf16* Kp = Kb + ((size_t)(b * NKV + kvh) * TT) * HEAD;
  const __bf16* Vp = Vt + ((size_t)(b * NKV + kvh) * HEAD) * TT;
  __bf16* Pw = Pb + wave * 16 * PSTR;

  // Q A-fragments (d=0..31 and d=32..63), resident for the whole loop
  FragU qa[2];
  #pragma unroll
  for (int c = 0; c < 2; ++c)
    #pragma unroll
    for (int j = 0; j < 8; ++j)
      qa[c].u[j] = *(const unsigned int*)(Q + (size_t)(q0 + n16) * HEAD + c * 32 + a_koff(j, hi));

  v8f o[4];
  #pragma unroll
  for (int f = 0; f < 4; ++f)
    for (int r = 0; r < 8; ++r) o[f][r] = 0.0f;
  float mrow[8], lrow[8];
  #pragma unroll
  for (int r = 0; r < 8; ++r) { mrow[r] = -3.0e38f; lrow[r] = 0.0f; }

  const float SC = 0.18033688011112042f;        // (1/sqrt(64)) * log2(e)

  const int nkc = (q0 + 47) >> 5;               // = ceil((q0+16)/32)
  for (int kc = 0; kc < nkc; ++kc) {
    const int kk = kc << 5;
    // S tiles: S = Q * K^T  (B-frag: lane n = key column, K-dim = head dim)
    v8f s[2];
    #pragma unroll
    for (int t2 = 0; t2 < 2; ++t2)
      for (int r = 0; r < 8; ++r) s[t2][r] = 0.0f;
    #pragma unroll
    for (int t2 = 0; t2 < 2; ++t2) {
      const int kb0 = kk + t2 * 16;
      #pragma unroll
      for (int c = 0; c < 2; ++c) {
        FragU kf;
        #pragma unroll
        for (int j = 0; j < 8; ++j)
          kf.u[j] = *(const unsigned int*)(Kp + (size_t)(kb0 + n16) * HEAD + c * 32 + hi * 16 + 2 * j);
        s[t2] = __builtin_amdgcn_wmma_f32_16x16x32_bf16(
            false, qa[c].v, false, kf.v, (short)0, s[t2], false, false);
      }
    }
    // online softmax (log2 domain); row r+8*hi lives across 16-lane half-wave
    #pragma unroll
    for (int r = 0; r < 8; ++r) {
      const int row = q0 + r + 8 * hi;
      float x0 = (kk + n16      <= row) ? s[0][r] * SC : -3.0e38f;
      float x1 = (kk + 16 + n16 <= row) ? s[1][r] * SC : -3.0e38f;
      float rm = fmaxf(x0, x1);
      #pragma unroll
      for (int off = 1; off < 16; off <<= 1)
        rm = fmaxf(rm, __shfl_xor(rm, off, 32));
      float mn = fmaxf(mrow[r], rm);
      float alpha = exp2f(mrow[r] - mn);
      float p0 = exp2f(x0 - mn);
      float p1 = exp2f(x1 - mn);
      float rs = p0 + p1;
      #pragma unroll
      for (int off = 1; off < 16; off <<= 1)
        rs += __shfl_xor(rs, off, 32);
      lrow[r] = lrow[r] * alpha + rs;
      mrow[r] = mn;
      #pragma unroll
      for (int f = 0; f < 4; ++f) o[f][r] *= alpha;
      // relayout P: C-layout -> LDS row-major [16 rows][32 keys]
      Pw[(r + 8 * hi) * PSTR + n16]      = (__bf16)p0;
      Pw[(r + 8 * hi) * PSTR + 16 + n16] = (__bf16)p1;
    }
    // same-wave DS ops are in-order; fence keeps the compiler from reordering
    asm volatile("s_wait_dscnt 0" ::: "memory");
    FragU pa;
    #pragma unroll
    for (int j = 0; j < 8; ++j)
      pa.u[j] = *(const unsigned int*)&Pw[n16 * PSTR + a_koff(j, hi)];
    // O += P * V   (B-frag from transposed V: contiguous key pairs)
    #pragma unroll
    for (int f = 0; f < 4; ++f) {
      FragU vb;
      #pragma unroll
      for (int j = 0; j < 8; ++j)
        vb.u[j] = *(const unsigned int*)(Vp + (size_t)(f * 16 + n16) * TT + kk + hi * 16 + 2 * j);
      o[f] = __builtin_amdgcn_wmma_f32_16x16x32_bf16(
          false, pa.v, false, vb.v, (short)0, o[f], false, false);
    }
    asm volatile("s_wait_dscnt 0" ::: "memory");
  }
  // normalize and write Y in [b, t, h*64+d] layout for the output GEMM
  #pragma unroll
  for (int f = 0; f < 4; ++f)
    #pragma unroll
    for (int r = 0; r < 8; ++r) {
      const int row = q0 + r + 8 * hi;
      float val = o[f][r] / lrow[r];
      Y[((size_t)b * TT + row) * EMB + h * HEAD + f * 16 + n16] = (__bf16)val;
    }
}

// ---------------------------------------------------------------- launch
extern "C" void kernel_launch(void* const* d_in, const int* in_sizes, int n_in,
                              void* d_out, int out_size, void* d_ws, size_t ws_size,
                              hipStream_t stream)
{
  (void)in_sizes; (void)n_in; (void)out_size; (void)ws_size;
  const float* x  = (const float*)d_in[0];
  const float* Wq = (const float*)d_in[1];
  const float* bq = (const float*)d_in[2];
  const float* Wk = (const float*)d_in[3];
  const float* bk = (const float*)d_in[4];
  const float* Wv = (const float*)d_in[5];
  const float* bv = (const float*)d_in[6];
  const float* Wo = (const float*)d_in[7];
  const float* bo = (const float*)d_in[8];

  char* base = (char*)d_ws;
  size_t off = 0;
  auto take = [&](size_t bytes) -> void* {
    void* r = base + off;
    off = (off + bytes + 255) & ~(size_t)255;
    return r;
  };
  __bf16* xb  = (__bf16*)take((size_t)MTOT * EMB * 2);
  __bf16* Wqb = (__bf16*)take((size_t)EMB * EMB * 2);
  __bf16* Wkb = (__bf16*)take((size_t)EMB * KVDIM * 2);
  __bf16* Wvb = (__bf16*)take((size_t)EMB * KVDIM * 2);
  __bf16* Wob = (__bf16*)take((size_t)EMB * EMB * 2);
  float*  qf  = (float*) take((size_t)MTOT * EMB * 4);
  float*  kf  = (float*) take((size_t)MTOT * KVDIM * 4);
  float*  vf  = (float*) take((size_t)MTOT * KVDIM * 4);
  __bf16* qbb = (__bf16*)take((size_t)MTOT * EMB * 2);
  __bf16* kbb = (__bf16*)take((size_t)MTOT * KVDIM * 2);
  __bf16* vtb = (__bf16*)take((size_t)MTOT * KVDIM * 2);
  __bf16* yb  = (__bf16*)qf;   // qf is dead after RoPE; reuse its 32MB for Y

  const int THR = 256;
  int n;
  n = MTOT * EMB;   cast_f32_bf16<<<(n + THR - 1) / THR, THR, 0, stream>>>(x,  xb,  n);
  n = EMB * EMB;    cast_f32_bf16<<<(n + THR - 1) / THR, THR, 0, stream>>>(Wq, Wqb, n);
  n = EMB * KVDIM;  cast_f32_bf16<<<(n + THR - 1) / THR, THR, 0, stream>>>(Wk, Wkb, n);
  n = EMB * KVDIM;  cast_f32_bf16<<<(n + THR - 1) / THR, THR, 0, stream>>>(Wv, Wvb, n);
  n = EMB * EMB;    cast_f32_bf16<<<(n + THR - 1) / THR, THR, 0, stream>>>(Wo, Wob, n);

  dim3 gq(EMB / GBN, MTOT / GBM);
  dim3 gkv(KVDIM / GBN, MTOT / GBM);
  gemm_bias_bf16<<<gq,  THR, 0, stream>>>(xb, Wqb, bq, qf, MTOT, EMB,   EMB);
  gemm_bias_bf16<<<gkv, THR, 0, stream>>>(xb, Wkb, bk, kf, MTOT, KVDIM, EMB);
  gemm_bias_bf16<<<gkv, THR, 0, stream>>>(xb, Wvb, bv, vf, MTOT, KVDIM, EMB);

  n = MTOT * EMB;   rope_to_heads<<<(n + THR - 1) / THR, THR, 0, stream>>>(qf, qbb, NQ,  n);
  n = MTOT * KVDIM; rope_to_heads<<<(n + THR - 1) / THR, THR, 0, stream>>>(kf, kbb, NKV, n);
  n = MTOT * KVDIM; v_cast_T<<<(n + THR - 1) / THR, THR, 0, stream>>>(vf, vtb, n);

  flash_attn<<<dim3(TT / 16 / 8, NQ, BB), THR, 0, stream>>>(qbb, kbb, vtb, yb);

  gemm_bias_bf16<<<gq, THR, 0, stream>>>(yb, Wob, bo, (float*)d_out, MTOT, EMB, EMB);
}
